// IAF_module_80745385165020
// MI455X (gfx1250) — matrix operand order repaired and checked by reading, not compiled, not run on hardware
//
#include <hip/hip_runtime.h>
#include <hip/hip_bf16.h>

typedef _Float16 h16 __attribute__((ext_vector_type(16)));
typedef _Float16 h8  __attribute__((ext_vector_type(8)));
typedef float    f8  __attribute__((ext_vector_type(8)));
typedef int      v4i __attribute__((vector_size(16)));

#define NPIX 100352   // 8*112*112
#define CCH  384
#define HW2D 12544    // 112*112
#define NWIN 2048     // 8*16*16
#define NHEAD 8
#define DH   48
#define WS2  49
#define ATT_SCALE 0.14433756729740643f  // 48^-0.5

// ---------------------------------------------------------------------------
// WMMA fragment helpers (CDNA5 16x16x32 f16, wave32)
// A (16x32, MxK): lane L<16 holds row L, K chunks {0..7, 16..23};
//                 lane L>=16 holds row L-16, K chunks {8..15, 24..31}.
// B (32x16, KxN) loaded from an NxK row-major view: lane L holds col (L&15)
//                 = row (L&15) of the NxK view, K chunks as A: {0..7, 16..23} / {8..15, 24..31}.
// C/D f32: VGPR r, lane L -> (M = r + 8*(L>>4), N = L&15).
// ---------------------------------------------------------------------------
__device__ inline f8 f8zero() {
  f8 z;
#pragma unroll
  for (int i = 0; i < 8; ++i) z[i] = 0.f;
  return z;
}

__device__ inline h16 frag_a(const _Float16* base, int stride) {
  int lane = threadIdx.x & 31;
  const _Float16* p = base + (lane & 15) * stride + ((lane >> 4) << 3);
  h8 lo = *(const h8*)(p);
  h8 hi = *(const h8*)(p + 16);
  h16 r;
#pragma unroll
  for (int i = 0; i < 8; ++i) { r[i] = lo[i]; r[i + 8] = hi[i]; }
  return r;
}

__device__ inline h16 frag_b(const _Float16* base, int stride) {
  int lane = threadIdx.x & 31;
  const _Float16* p = base + (lane & 15) * stride + ((lane >> 4) << 3);
  h8 lo = *(const h8*)(p);
  h8 hi = *(const h8*)(p + 16);
  h16 r;
#pragma unroll
  for (int i = 0; i < 8; ++i) { r[i] = lo[i]; r[i + 8] = hi[i]; }
  return r;
}

__device__ inline f8 wmma32(h16 a, h16 b, f8 c) {
  // D = A*B + C, fp32 accum, 16x16x32 f16
  return __builtin_amdgcn_wmma_f32_16x16x32_f16(false, a, false, b, (short)0, c,
                                                false, false);
}

// ---------------------------------------------------------------------------
// Async global -> LDS copy (CDNA5): GLOBAL_LOAD_ASYNC_TO_LDS_B128, ASYNCcnt.
// Builtin param 0 is a (non-const) v4i pointer in address_space(1) per the
// compiler diagnostic; param 1 is the LDS-side pointer (address_space(3)).
// ---------------------------------------------------------------------------
__device__ inline void async_copy16(const _Float16* g, _Float16* l) {
#if __has_builtin(__builtin_amdgcn_global_load_async_to_lds_b128)
  __builtin_amdgcn_global_load_async_to_lds_b128(
      (__attribute__((address_space(1))) v4i*)g,
      (__attribute__((address_space(3))) v4i*)l, 0, 0);
#else
  *(h8*)l = *(const h8*)g;
#endif
}

template <int N>
__device__ inline void wait_async() {
#if __has_builtin(__builtin_amdgcn_s_wait_asynccnt)
  __builtin_amdgcn_s_wait_asynccnt(N);
#else
  asm volatile("s_wait_asynccnt %0" ::"i"(N) : "memory");
#endif
}

// window-order row -> linear pixel row (b*12544 + h*112 + w)
__device__ inline int worder_to_linear(int m) {
  int win = m / WS2, t = m % WS2;
  int b  = win >> 8;
  int i1 = (win >> 4) & 15;
  int i2 = win & 15;
  int h1 = t / 7, w1 = t % 7;
  int h = h1 * 16 + i1, w = w1 * 16 + i2;
  return (b * 112 + h) * 112 + w;
}

// ---------------------------------------------------------------------------
// Prep kernels: fold BN to scale/bias, convert weights fp32 -> f16
// ---------------------------------------------------------------------------
__global__ __launch_bounds__(64) void fold_bn_kernel(
    const float* __restrict__ g, const float* __restrict__ b,
    const float* __restrict__ rm, const float* __restrict__ rv, int n,
    float* __restrict__ os, float* __restrict__ ob) {
  int i = blockIdx.x * 64 + threadIdx.x;
  if (i < n) {
    float s = g[i] * rsqrtf(rv[i] + 1e-5f);
    os[i] = s;
    ob[i] = b[i] - rm[i] * s;
  }
}

__global__ __launch_bounds__(256) void cvt_w_kernel(
    const float* __restrict__ qw, const float* __restrict__ kw,
    const float* __restrict__ vw, const float* __restrict__ pw,
    _Float16* __restrict__ out) {
  int idx = blockIdx.x * 256 + threadIdx.x;
  if (idx >= 4 * 147456) return;
  int which = idx / 147456, r = idx % 147456;
  const float* src = (which == 0) ? qw : (which == 1) ? kw : (which == 2) ? vw : pw;
  out[idx] = (_Float16)src[r];
}

// ---------------------------------------------------------------------------
// LayerNorm over channels; writes f16 activations in window order [win*49+t, C]
// ---------------------------------------------------------------------------
__global__ __launch_bounds__(256) void ln_kernel(const float* __restrict__ x,
                                                 const float* __restrict__ g,
                                                 const float* __restrict__ b,
                                                 _Float16* __restrict__ out) {
  int p = blockIdx.x * 256 + threadIdx.x;
  if (p >= NPIX) return;
  int bb = p / HW2D, hw = p % HW2D;
  int h = hw / 112, w = hw % 112;
  const float* xp = x + (size_t)bb * CCH * HW2D + hw;
  float sum = 0.f, sq = 0.f;
  for (int c = 0; c < CCH; ++c) {
    float v = xp[(size_t)c * HW2D];
    sum += v; sq += v * v;
  }
  float mean = sum * (1.f / CCH);
  float var  = sq * (1.f / CCH) - mean * mean;
  float rstd = rsqrtf(var + 1e-6f);
  int i1 = h & 15, h1 = h >> 4, i2 = w & 15, w1 = w >> 4;
  int row = ((bb * 16 + i1) * 16 + i2) * WS2 + (h1 * 7 + w1);
  _Float16* op = out + (size_t)row * CCH;
  for (int c = 0; c < CCH; ++c) {
    float v = xp[(size_t)c * HW2D];
    op[c] = (_Float16)((v - mean) * rstd * g[c] + b[c]);
  }
}

// ---------------------------------------------------------------------------
// GEMM: Y[m, o] = relu(scale[o] * sum_c X[m,c]*W[o,c] + bias[o])
// M = 100352, N = 384, K = 384.  Block tile 128x128, K-step 32, 8 waves.
// Double-buffered LDS with async global->LDS copies (one K-slab in flight).
// MODE 0: f16 out, rows identity (window order)
// MODE 1: f16 out, rows remapped window-order -> linear (NHWC)
// MODE 2: f32 out, NCHW into d_out
// ---------------------------------------------------------------------------
template <int MODE>
__global__ __launch_bounds__(256) void gemm_kernel(
    const _Float16* __restrict__ X, const _Float16* __restrict__ Wt,
    const float* __restrict__ sc, const float* __restrict__ bi,
    void* __restrict__ Yv) {
  __shared__ __align__(16) _Float16 Xs[2][128][40];
  __shared__ __align__(16) _Float16 Wsh[2][128][40];
  const int m0 = blockIdx.x * 128, n0 = blockIdx.y * 128;
  const int tid = threadIdx.x, lane = tid & 31, wave = tid >> 5;
  const int wm = wave & 3, wn = wave >> 2;  // 4 M-strips x 2 N-strips

  f8 acc[2][4];
#pragma unroll
  for (int i = 0; i < 2; ++i)
#pragma unroll
    for (int j = 0; j < 4; ++j) acc[i][j] = f8zero();

  // issue one K-slab (32 wide) into LDS stage `bi`; 4 async B128 per thread
  const int ld_r  = tid >> 2;
  const int ld_c8 = (tid & 3) * 8;
  auto issue_tile = [&](int bi, int k0) {
    async_copy16(&X[(size_t)(m0 + ld_r) * CCH + k0 + ld_c8], &Xs[bi][ld_r][ld_c8]);
    async_copy16(&Wt[(size_t)(n0 + ld_r) * CCH + k0 + ld_c8], &Wsh[bi][ld_r][ld_c8]);
    async_copy16(&X[(size_t)(m0 + 64 + ld_r) * CCH + k0 + ld_c8], &Xs[bi][64 + ld_r][ld_c8]);
    async_copy16(&Wt[(size_t)(n0 + 64 + ld_r) * CCH + k0 + ld_c8], &Wsh[bi][64 + ld_r][ld_c8]);
  };

  issue_tile(0, 0);
  for (int s = 0; s < 12; ++s) {
    const int cur = s & 1;
    if (s + 1 < 12) {
      issue_tile(cur ^ 1, (s + 1) * 32);
      wait_async<4>();  // current slab's 4 copies complete; next 4 in flight
    } else {
      wait_async<0>();
    }
    __syncthreads();

    h16 afr[2], bfr[4];
#pragma unroll
    for (int i = 0; i < 2; ++i)
      afr[i] = frag_a(&Xs[cur][wm * 32 + i * 16][0], 40);
#pragma unroll
    for (int j = 0; j < 4; ++j)
      bfr[j] = frag_b(&Wsh[cur][wn * 64 + j * 16][0], 40);
#pragma unroll
    for (int i = 0; i < 2; ++i)
#pragma unroll
      for (int j = 0; j < 4; ++j) acc[i][j] = wmma32(afr[i], bfr[j], acc[i][j]);
    __syncthreads();  // everyone done reading `cur` before it is refilled
  }

  const int l15 = lane & 15, lhi = lane >> 4;
#pragma unroll
  for (int j = 0; j < 4; ++j) {
    int o = n0 + wn * 64 + j * 16 + l15;
    float s_ = sc[o], b_ = bi[o];
#pragma unroll
    for (int i = 0; i < 2; ++i) {
#pragma unroll
      for (int r = 0; r < 8; ++r) {
        int m = m0 + wm * 32 + i * 16 + r + lhi * 8;
        float v = fmaxf(acc[i][j][r] * s_ + b_, 0.f);
        if (MODE == 0) {
          ((_Float16*)Yv)[(size_t)m * CCH + o] = (_Float16)v;
        } else if (MODE == 1) {
          int row = worder_to_linear(m);
          ((_Float16*)Yv)[(size_t)row * CCH + o] = (_Float16)v;
        } else {
          int b2 = m / HW2D, hw = m % HW2D;
          ((float*)Yv)[((size_t)b2 * CCH + o) * HW2D + hw] = v;
        }
      }
    }
  }
}

// ---------------------------------------------------------------------------
// Attention per (window, head): S = Q K^T * scale, softmax, O = P V, plus the
// fused depthwise-3x3 over the (pos=49, dim=48) plane on V. 4 waves/block,
// each wave owns one 16-row strip of the padded 64x64 problem.
// ---------------------------------------------------------------------------
__global__ __launch_bounds__(128) void attn_kernel(
    const _Float16* __restrict__ Q, const _Float16* __restrict__ K,
    const _Float16* __restrict__ V, const float* __restrict__ dwcw,
    const float* __restrict__ sb, _Float16* __restrict__ Xspa) {
  __shared__ __align__(16) _Float16 Qs[64][72];
  __shared__ __align__(16) _Float16 Ks[64][72];
  __shared__ __align__(16) _Float16 Vt[48][72];  // transposed: [dim][pos]
  __shared__ __align__(16) _Float16 Ps[64][72];

  const int win = blockIdx.x, head = blockIdx.y;
  const int tid = threadIdx.x, lane = tid & 31, wave = tid >> 5;
  const size_t base = (size_t)win * WS2 * CCH + head * DH;

  for (int idx = tid; idx < 64 * 64; idx += 128) {
    int t = idx >> 6, d = idx & 63;
    _Float16 qv = (_Float16)0.f, kv = (_Float16)0.f;
    if (t < WS2 && d < DH) {
      qv = Q[base + (size_t)t * CCH + d];
      kv = K[base + (size_t)t * CCH + d];
    }
    Qs[t][d] = qv;
    Ks[t][d] = kv;
  }
  for (int idx = tid; idx < 48 * 64; idx += 128) {
    int d = idx >> 6, t = idx & 63;
    Vt[d][t] = (t < WS2) ? V[base + (size_t)t * CCH + d] : (_Float16)0.f;
  }
  __syncthreads();

  // --- S = Q K^T (wave strip: rows [wave*16, wave*16+16), all 64 cols) ---
  f8 accS[4];
#pragma unroll
  for (int j = 0; j < 4; ++j) accS[j] = f8zero();
#pragma unroll
  for (int s = 0; s < 2; ++s) {
    h16 a = frag_a(&Qs[wave * 16][s * 32], 72);
#pragma unroll
    for (int j = 0; j < 4; ++j) {
      h16 b = frag_b(&Ks[j * 16][s * 32], 72);
      accS[j] = wmma32(a, b, accS[j]);
    }
  }

  // --- row softmax (masking n >= 49), write P to LDS in f16 ---
  const int l15 = lane & 15, lhi = lane >> 4;
#pragma unroll
  for (int r = 0; r < 8; ++r) {
    float vj[4];
    float mx = -1e30f;
#pragma unroll
    for (int j = 0; j < 4; ++j) {
      int n = j * 16 + l15;
      vj[j] = (n < WS2) ? accS[j][r] * ATT_SCALE : -1e30f;
      mx = fmaxf(mx, vj[j]);
    }
#pragma unroll
    for (int msk = 1; msk < 16; msk <<= 1) mx = fmaxf(mx, __shfl_xor(mx, msk, 32));
    float ej[4], sum = 0.f;
#pragma unroll
    for (int j = 0; j < 4; ++j) {
      int n = j * 16 + l15;
      ej[j] = (n < WS2) ? __expf(vj[j] - mx) : 0.f;
      sum += ej[j];
    }
#pragma unroll
    for (int msk = 1; msk < 16; msk <<= 1) sum += __shfl_xor(sum, msk, 32);
    float inv = 1.f / sum;
    int mrow = wave * 16 + r + lhi * 8;
#pragma unroll
    for (int j = 0; j < 4; ++j) Ps[mrow][j * 16 + l15] = (_Float16)(ej[j] * inv);
  }
  __syncthreads();

  // --- O = P V  (N = 48 dims, K = 64 padded positions) ---
  f8 accO[3];
#pragma unroll
  for (int j = 0; j < 3; ++j) accO[j] = f8zero();
#pragma unroll
  for (int s = 0; s < 2; ++s) {
    h16 a = frag_a(&Ps[wave * 16][s * 32], 72);
#pragma unroll
    for (int j = 0; j < 3; ++j) {
      h16 b = frag_b(&Vt[j * 16][s * 32], 72);
      accO[j] = wmma32(a, b, accO[j]);
    }
  }

  // --- fused depthwise 3x3 over (pos, dim) on V + BN + ReLU, then store ---
  float w9[9];
#pragma unroll
  for (int i = 0; i < 9; ++i) w9[i] = dwcw[head * 9 + i];
  float dsc = sb[3072 + head], dbi = sb[3080 + head];
#pragma unroll
  for (int j = 0; j < 3; ++j) {
#pragma unroll
    for (int r = 0; r < 8; ++r) {
      int m = wave * 16 + r + lhi * 8;
      if (m < WS2) {
        int d = j * 16 + l15;
        float conv = 0.f;
#pragma unroll
        for (int dy = 0; dy < 3; ++dy)
#pragma unroll
          for (int dx = 0; dx < 3; ++dx) {
            int mm = m + dy - 1, dd = d + dx - 1;
            if (mm >= 0 && mm < WS2 && dd >= 0 && dd < DH)
              conv += w9[dy * 3 + dx] * (float)Vt[dd][mm];
          }
        float dv = fmaxf(conv * dsc + dbi, 0.f);
        Xspa[base + (size_t)m * CCH + d] = (_Float16)(accO[j][r] + dv);
      }
    }
  }
}

// ---------------------------------------------------------------------------
// Spatial depthwise 3x3 + BN + ReLU + skip, NHWC f16 in/out
// ---------------------------------------------------------------------------
__global__ __launch_bounds__(256) void loc_kernel(const _Float16* __restrict__ U,
                                                  const float* __restrict__ lw,
                                                  const float* __restrict__ sb,
                                                  _Float16* __restrict__ Z) {
  int idx = blockIdx.x * 256 + threadIdx.x;  // NPIX*CCH total, exact multiple
  int c = idx % CCH;
  int p = idx / CCH;
  int bb = p / HW2D, hw = p % HW2D;
  int h = hw / 112, w = hw % 112;
  const float* lwp = lw + c * 9;
  float conv = 0.f;
#pragma unroll
  for (int dy = 0; dy < 3; ++dy)
#pragma unroll
    for (int dx = 0; dx < 3; ++dx) {
      int hh = h + dy - 1, ww = w + dx - 1;
      if (hh >= 0 && hh < 112 && ww >= 0 && ww < 112)
        conv += lwp[dy * 3 + dx] *
                (float)U[(((size_t)bb * 112 + hh) * 112 + ww) * CCH + c];
    }
  float s_ = sb[3088 + c], b_ = sb[3472 + c];
  float u = (float)U[(size_t)idx];
  Z[idx] = (_Float16)(u + fmaxf(conv * s_ + b_, 0.f));
}

// ---------------------------------------------------------------------------
// Host-side orchestration
// ---------------------------------------------------------------------------
extern "C" void kernel_launch(void* const* d_in, const int* in_sizes, int n_in,
                              void* d_out, int out_size, void* d_ws, size_t ws_size,
                              hipStream_t stream) {
  const float* x1   = (const float*)d_in[0];
  const float* x2   = (const float*)d_in[1];
  const float* ln_g = (const float*)d_in[2];
  const float* ln_b = (const float*)d_in[3];
  const float* q_w  = (const float*)d_in[4];
  const float* k_w  = (const float*)d_in[9];
  const float* v_w  = (const float*)d_in[14];
  const float* p_w  = (const float*)d_in[19];
  const float* dwc_w = (const float*)d_in[24];
  const float* loc_w = (const float*)d_in[29];

  const size_t RBYTES = (size_t)NPIX * CCH * sizeof(_Float16);  // 77,070,336
  char* ws = (char*)d_ws;
  _Float16* R0 = (_Float16*)(ws);               // x1_ln -> later x_spa
  _Float16* R1 = (_Float16*)(ws + RBYTES);      // x2_ln (= V, lives long)
  _Float16* R2 = (_Float16*)(ws + 2 * RBYTES);  // Q -> later Z
  _Float16* R3 = (_Float16*)(ws + 3 * RBYTES);  // K -> later U
  _Float16* WH = (_Float16*)(ws + 4 * RBYTES);  // 4 f16 weight matrices
  float*    SB = (float*)(ws + 4 * RBYTES + 4 * 147456 * sizeof(_Float16));

  // --- prep: fold BN params, convert weights ---
  cvt_w_kernel<<<(4 * 147456 + 255) / 256, 256, 0, stream>>>(q_w, k_w, v_w, p_w, WH);
  fold_bn_kernel<<<6, 64, 0, stream>>>((const float*)d_in[5],  (const float*)d_in[6],
                                       (const float*)d_in[7],  (const float*)d_in[8],  384, SB + 0,    SB + 384);
  fold_bn_kernel<<<6, 64, 0, stream>>>((const float*)d_in[10], (const float*)d_in[11],
                                       (const float*)d_in[12], (const float*)d_in[13], 384, SB + 768,  SB + 1152);
  fold_bn_kernel<<<6, 64, 0, stream>>>((const float*)d_in[15], (const float*)d_in[16],
                                       (const float*)d_in[17], (const float*)d_in[18], 384, SB + 1536, SB + 1920);
  fold_bn_kernel<<<6, 64, 0, stream>>>((const float*)d_in[20], (const float*)d_in[21],
                                       (const float*)d_in[22], (const float*)d_in[23], 384, SB + 2304, SB + 2688);
  fold_bn_kernel<<<1, 64, 0, stream>>>((const float*)d_in[25], (const float*)d_in[26],
                                       (const float*)d_in[27], (const float*)d_in[28], 8,   SB + 3072, SB + 3080);
  fold_bn_kernel<<<6, 64, 0, stream>>>((const float*)d_in[30], (const float*)d_in[31],
                                       (const float*)d_in[32], (const float*)d_in[33], 384, SB + 3088, SB + 3472);

  // --- LayerNorm both inputs -> f16 window-order activations ---
  ln_kernel<<<NPIX / 256, 256, 0, stream>>>(x1, ln_g, ln_b, R0);
  ln_kernel<<<NPIX / 256, 256, 0, stream>>>(x2, ln_g, ln_b, R1);

  dim3 gg(NPIX / 128, CCH / 128);  // 784 x 3

  // Q = relu(bn_q(x1_ln @ q_w^T)), K = relu(bn_k(x2_ln @ k_w^T))
  gemm_kernel<0><<<gg, 256, 0, stream>>>(R0, WH + 0 * 147456, SB + 0,   SB + 384,  (void*)R2);
  gemm_kernel<0><<<gg, 256, 0, stream>>>(R1, WH + 1 * 147456, SB + 768, SB + 1152, (void*)R3);

  // attention + fused dwc -> x_spa (window order, f16) into R0
  attn_kernel<<<dim3(NWIN, NHEAD), 128, 0, stream>>>(R2, R3, R1, dwc_w, SB, R0);

  // v-projection: x_spa -> U (NHWC f16, linear pixel rows) into R3
  gemm_kernel<1><<<gg, 256, 0, stream>>>(R0, WH + 2 * 147456, SB + 1536, SB + 1920, (void*)R3);

  // local depthwise 3x3 + skip -> Z (NHWC f16) into R2
  loc_kernel<<<(NPIX * CCH) / 256, 256, 0, stream>>>(R3, loc_w, SB, R2);

  // p-projection -> fp32 NCHW output
  gemm_kernel<2><<<gg, 256, 0, stream>>>(R2, WH + 3 * 147456, SB + 2304, SB + 2688, d_out);
}
